// MHSA_37185826849009
// MI455X (gfx1250) — matrix operand order repaired and checked by reading, not compiled, not run on hardware
//
#include <hip/hip_runtime.h>
#include <hip/hip_bf16.h>

#define D_MODEL 1024
#define N_HEADS 16
#define D_HEAD  64
#define BATCH   4
#define SEQ     2048

typedef __bf16 bf16_t;
typedef __attribute__((ext_vector_type(16))) __bf16 v16bf;
typedef __attribute__((ext_vector_type(8)))  __bf16 v8bf;
typedef __attribute__((ext_vector_type(8)))  float  v8f;
typedef __attribute__((ext_vector_type(4)))  unsigned int u32x4;
typedef __attribute__((ext_vector_type(8)))  int i32x8;
typedef __attribute__((ext_vector_type(4)))  int i32x4;

// native RNE convert (v_cvt_*bf16_f32) instead of manual integer rounding
__device__ __forceinline__ bf16_t f2bf(float f) { return (bf16_t)f; }

__device__ __forceinline__ v8f wmma_bf16(v16bf a, v16bf b, v8f c) {
    return __builtin_amdgcn_wmma_f32_16x16x32_bf16(
        false, a, false, b, (short)0, c, false, false);
}

union V16U { v16bf v; v8bf h[2]; };

// A-fragment (16x32 bf16): lane holds row (lane&15); elements j = K
// b0+j, b0 = 16*(lane>=16).
__device__ __forceinline__ v16bf load_afrag(const bf16_t* p, int rs, int lane) {
    const int row = lane & 15, b0 = (lane >> 4) * 16;
    V16U u;
    u.h[0] = *(const v8bf*)(p + row * rs + b0);
    u.h[1] = *(const v8bf*)(p + row * rs + b0 + 8);
    return u.v;
}

// B-fragment (32x16 bf16) from a [N][K]-major tile: lane holds column
// (lane&15); elements j = K (j + 16*(lane>=16)) -> one contiguous 32B read.
__device__ __forceinline__ v16bf load_bfrag(const bf16_t* p, int rs, int lane) {
    const int nn = lane & 15, hi = lane >> 4;
    return *(const v16bf*)(p + nn * rs + hi * 16);
}

__device__ __forceinline__ float rmax16(float v) {
#pragma unroll
    for (int m = 8; m >= 1; m >>= 1) v = fmaxf(v, __shfl_xor(v, m, 16));
    return v;
}
__device__ __forceinline__ float rsum16(float v) {
#pragma unroll
    for (int m = 8; m >= 1; m >>= 1) v += __shfl_xor(v, m, 16);
    return v;
}

// TDM: 2D bf16 tile load global -> LDS. D# per CDNA5 ISA.
// amdgpu-toolchain (clang-23) 6-arg builtin: (g0, g1, g2, g3, extra i32x8, cpol)
__device__ __forceinline__ void tdm_load_2d_bf16(
    unsigned lds_off, const void* gaddr,
    unsigned tensor_w, unsigned tensor_h, unsigned row_stride,
    unsigned tile_w, unsigned tile_h)
{
    unsigned long long ga = (unsigned long long)(size_t)gaddr;
    u32x4 g0 = { 1u,                                   // count=1 (user D#)
                 lds_off,                              // lds_addr (bytes)
                 (unsigned)(ga & 0xffffffffu),         // global_addr lo
                 (unsigned)((ga >> 32) & 0x01ffffffu)  // global_addr hi
                     | (2u << 30) };                   // type=2 ("image")
    i32x8 g1 = { (int)(1u << 16),                                  // data_size=1 -> 2B
                 (int)((tensor_w & 0xffffu) << 16),                // dim0[15:0]
                 (int)((tensor_w >> 16) | ((tensor_h & 0xffffu) << 16)), // dim0 hi | dim1 lo
                 (int)((tensor_h >> 16) | (tile_w << 16)),         // dim1 hi | tile_dim0
                 (int)tile_h,                                      // tile_dim1 | tile_dim2=0
                 (int)row_stride,                                  // dim0_stride lo32
                 0, 0 };
    i32x4 g2 = { 0, 0, 0, 0 };
    i32x4 g3 = { 0, 0, 0, 0 };
    i32x8 g4 = { 0, 0, 0, 0, 0, 0, 0, 0 };             // unused trailing group
    __builtin_amdgcn_tensor_load_to_lds(g0, g1, g2, g3, g4, 0);
}

// ---------------------------------------------------------------------------
// Kernel 1: QKV projection (f32 GEMM inputs -> bf16 heads). V is emitted
// TRANSPOSED per head (vt[b][h][dh][s]) so attention's PV B-fragments become
// contiguous global loads (no LDS transpose / no block barriers in attention).
// ---------------------------------------------------------------------------
__global__ __launch_bounds__(128) void qkv_proj_kernel(
    const float* __restrict__ xq, const float* __restrict__ xk, const float* __restrict__ xv,
    const float* __restrict__ Wq, const float* __restrict__ Wk, const float* __restrict__ Wv,
    const float* __restrict__ bq, const float* __restrict__ bk, const float* __restrict__ bv,
    bf16_t* __restrict__ oq, bf16_t* __restrict__ ok, bf16_t* __restrict__ ov)
{
    const float* X; const float* W; const float* bias; bf16_t* out;
    if (blockIdx.z == 0)      { X = xq; W = Wq; bias = bq; out = oq; }
    else if (blockIdx.z == 1) { X = xk; W = Wk; bias = bk; out = ok; }
    else                      { X = xv; W = Wv; bias = bv; out = ov; }

    const int m0 = blockIdx.x * 64, n0 = blockIdx.y * 64;
    const int tid = threadIdx.x, wave = tid >> 5, lane = tid & 31;
    const int nn = lane & 15, hi = lane >> 4;

    __shared__ bf16_t Xs[64 * 32];   // [m][k]
    __shared__ bf16_t Ws[64 * 32];   // [n][k]

    const v8f z = {0.f,0.f,0.f,0.f,0.f,0.f,0.f,0.f};
    v8f acc[4]; acc[0]=z; acc[1]=z; acc[2]=z; acc[3]=z;

    for (int kk = 0; kk < D_MODEL; kk += 32) {
        {
            const int row = tid >> 1, k0 = (tid & 1) * 16;
            const float* src = X + (size_t)(m0 + row) * D_MODEL + kk + k0;
#pragma unroll
            for (int j = 0; j < 16; ++j) Xs[row * 32 + k0 + j] = f2bf(src[j]);
        }
        {
            const int kr = tid >> 2, nq = (tid & 3) * 16;
            const float* src = W + (size_t)(kk + kr) * D_MODEL + n0 + nq;
#pragma unroll
            for (int j = 0; j < 16; ++j) Ws[(nq + j) * 32 + kr] = f2bf(src[j]);
        }
        __syncthreads();
        v16bf a = load_afrag(Xs + wave * 16 * 32, 32, lane);
#pragma unroll
        for (int nt = 0; nt < 4; ++nt)
            acc[nt] = wmma_bf16(a, load_bfrag(Ws + nt * 16 * 32, 32, lane), acc[nt]);
        __syncthreads();
    }

#pragma unroll
    for (int nt = 0; nt < 4; ++nt) {
        const int col = n0 + nt * 16 + nn;
        const float bb = bias[col];
        const int hh = col >> 6, dh = col & 63;
#pragma unroll
        for (int r = 0; r < 8; ++r) {
            const int row = m0 + wave * 16 + r + 8 * hi;
            const int b = row >> 11, s = row & 2047;
            const bf16_t val = f2bf(acc[nt][r] + bb);
            if (blockIdx.z == 2)  // V: [b][h][dh][s] (transposed)
                out[(((size_t)b * N_HEADS + hh) * D_HEAD + dh) * SEQ + s] = val;
            else                  // Q,K: [b][h][s][dh]
                out[(((size_t)b * N_HEADS + hh) * SEQ + s) * D_HEAD + dh] = val;
        }
    }
}

// ---------------------------------------------------------------------------
// Kernel 2: flash attention, barrier-free. Q/K fragments direct from global;
// V^T fragments direct from global; P re-shaped via per-wave LDS ordered by
// s_wait_dscnt. Online softmax uses 16-lane shuffles matching the C layout.
// ---------------------------------------------------------------------------
__global__ __launch_bounds__(128) void attn_kernel(
    const bf16_t* __restrict__ qh, const bf16_t* __restrict__ kh,
    const bf16_t* __restrict__ vt, const float* __restrict__ beta,
    const int* __restrict__ mask, bf16_t* __restrict__ attn_out)
{
    const int bh = blockIdx.x;                // 0..63
    const int b = bh >> 4, h = bh & 15;
    const int tid = threadIdx.x, wave = tid >> 5, lane = tid & 31;
    const int nn = lane & 15, hi = lane >> 4;

    __shared__ bf16_t Pl[4][16 * 32];         // per-wave P tile [m][k]

    const size_t bhoff = (size_t)bh * SEQ * D_HEAD;
    const bf16_t* qb  = qh + bhoff;
    const bf16_t* kb  = kh + bhoff;
    const bf16_t* vtb = vt + bhoff;           // [dh][s] per head

    const int q0 = blockIdx.y * 64 + wave * 16;

    v16bf aq0 = load_afrag(qb + (size_t)q0 * D_HEAD + 0,  D_HEAD, lane);
    v16bf aq1 = load_afrag(qb + (size_t)q0 * D_HEAD + 32, D_HEAD, lane);

    const v8f z = {0.f,0.f,0.f,0.f,0.f,0.f,0.f,0.f};
    v8f o[4]; o[0]=z; o[1]=z; o[2]=z; o[3]=z;
    float mi[8], li[8];
#pragma unroll
    for (int r = 0; r < 8; ++r) { mi[r] = -3.0e38f; li[r] = 0.f; }

    const int*   mb    = mask + b * SEQ;
    const float* betah = beta + (size_t)h * SEQ * SEQ;

    for (int kc = 0; kc < SEQ; kc += 32) {
        if (kc + 32 < SEQ) {  // global_prefetch_b8 next K chunk
            __builtin_prefetch(kb + (size_t)(kc + 32) * D_HEAD, 0, 1);
        }

        // scores S[16q x 32k]: K B-frags straight from HBM (row-major K)
        v8f s0 = z, s1 = z;
        {
            const bf16_t* kt0 = kb + (size_t)kc * D_HEAD;
            const bf16_t* kt1 = kt0 + 16 * D_HEAD;
            s0 = wmma_bf16(aq0, load_bfrag(kt0 + 0,  D_HEAD, lane), s0);
            s0 = wmma_bf16(aq1, load_bfrag(kt0 + 32, D_HEAD, lane), s0);
            s1 = wmma_bf16(aq0, load_bfrag(kt1 + 0,  D_HEAD, lane), s1);
            s1 = wmma_bf16(aq1, load_bfrag(kt1 + 32, D_HEAD, lane), s1);
        }

        const int c0 = kc + nn, c1 = kc + 16 + nn;
        const bool z0 = (mb[c0] == 0), z1 = (mb[c1] == 0);

        float alpha[8];
#pragma unroll
        for (int r = 0; r < 8; ++r) {
            const int qr = q0 + r + 8 * hi;
            const float* brow = betah + (size_t)qr * SEQ;
            float x0 = s0[r] * 0.125f + brow[c0];
            float x1 = s1[r] * 0.125f + brow[c1];
            if (z0) x0 = -1.0e9f;
            if (z1) x1 = -1.0e9f;
            const float cm = rmax16(fmaxf(x0, x1));
            const float mn = fmaxf(mi[r], cm);
            const float al = __expf(mi[r] - mn);
            const float p0 = __expf(x0 - mn);
            const float p1 = __expf(x1 - mn);
            li[r] = li[r] * al + rsum16(p0 + p1);
            mi[r] = mn; alpha[r] = al;
            const int m = r + 8 * hi;
            Pl[wave][m * 32 + nn]      = f2bf(p0);
            Pl[wave][m * 32 + 16 + nn] = f2bf(p1);
        }
#pragma unroll
        for (int nt = 0; nt < 4; ++nt)
#pragma unroll
            for (int r = 0; r < 8; ++r) o[nt][r] *= alpha[r];

        // own-wave LDS RAW: drain DScnt before re-reading P as an A-fragment
        asm volatile("s_wait_dscnt 0" ::: "memory");
        v16bf pa = load_afrag(&Pl[wave][0], 32, lane);
#pragma unroll
        for (int nt = 0; nt < 4; ++nt)   // V^T B-frags straight from HBM
            o[nt] = wmma_bf16(pa,
                load_bfrag(vtb + (size_t)(nt * 16) * SEQ + kc, SEQ, lane), o[nt]);
    }

    // normalize; store to attn workspace [B][S][D_MODEL] bf16
#pragma unroll
    for (int r = 0; r < 8; ++r) {
        const float inv = 1.0f / li[r];
        const int qr = q0 + r + 8 * hi;
        bf16_t* orow = attn_out + ((size_t)b * SEQ + qr) * D_MODEL + h * D_HEAD;
#pragma unroll
        for (int nt = 0; nt < 4; ++nt)
            orow[nt * 16 + nn] = f2bf(o[nt][r] * inv);
    }
}

// ---------------------------------------------------------------------------
// Kernel 3: output projection. A tile (bf16, row-major) is staged by the
// Tensor Data Mover, double-buffered: issue tile i+1, s_wait_tensorcnt 1
// guarantees tile i landed, then barrier and compute.
// ---------------------------------------------------------------------------
__global__ __launch_bounds__(128) void out_proj_kernel(
    const bf16_t* __restrict__ A, const float* __restrict__ W,
    const float* __restrict__ bias, float* __restrict__ out)
{
    const int m0 = blockIdx.x * 64, n0 = blockIdx.y * 64;
    const int tid = threadIdx.x, wave = tid >> 5, lane = tid & 31;
    const int nn = lane & 15, hi = lane >> 4;

    __shared__ bf16_t Xs[2][64 * 32];   // TDM double buffer
    __shared__ bf16_t Ws[64 * 32];      // [n][k]

    const unsigned xs_base = (unsigned)(size_t)(void*)&Xs[0][0]; // LDS byte offset

    const v8f z = {0.f,0.f,0.f,0.f,0.f,0.f,0.f,0.f};
    v8f acc[4]; acc[0]=z; acc[1]=z; acc[2]=z; acc[3]=z;

    // prologue: TDM fetch of tile kk=0 into buffer 0
    if (wave == 0)
        tdm_load_2d_bf16(xs_base, A + (size_t)m0 * D_MODEL,
                         D_MODEL, BATCH * SEQ, D_MODEL, 32, 64);

    for (int kk = 0, it = 0; kk < D_MODEL; kk += 32, ++it) {
        const int buf = it & 1;
        {   // stage W tile transposed (all threads)
            const int kr = tid >> 2, nq = (tid & 3) * 16;
            const float* src = W + (size_t)(kk + kr) * D_MODEL + n0 + nq;
#pragma unroll
            for (int j = 0; j < 16; ++j) Ws[(nq + j) * 32 + kr] = f2bf(src[j]);
        }
        if (wave == 0) {
            if (kk + 32 < D_MODEL) {
                tdm_load_2d_bf16(xs_base + (unsigned)((buf ^ 1) * 64 * 32 * 2),
                                 A + (size_t)m0 * D_MODEL + kk + 32,
                                 D_MODEL, BATCH * SEQ, D_MODEL, 32, 64);
                __builtin_amdgcn_s_wait_tensorcnt(1);  // current tile landed
            } else {
                __builtin_amdgcn_s_wait_tensorcnt(0);
            }
        }
        __syncthreads();   // Xs[buf] (tensor) + Ws (ds) visible to all waves

        v16bf a = load_afrag(Xs[buf] + wave * 16 * 32, 32, lane);
#pragma unroll
        for (int nt = 0; nt < 4; ++nt)
            acc[nt] = wmma_bf16(a, load_bfrag(Ws + nt * 16 * 32, 32, lane), acc[nt]);
        __syncthreads();   // reads done before Ws / other Xs buffer rewritten
    }

#pragma unroll
    for (int nt = 0; nt < 4; ++nt) {
        const int col = n0 + nt * 16 + nn;
        const float bb = bias[col];
#pragma unroll
        for (int r = 0; r < 8; ++r) {
            const int row = m0 + wave * 16 + r + 8 * hi;
            out[(size_t)row * D_MODEL + col] = acc[nt][r] + bb;
        }
    }
}

// ---------------------------------------------------------------------------
extern "C" void kernel_launch(void* const* d_in, const int* in_sizes, int n_in,
                              void* d_out, int out_size, void* d_ws, size_t ws_size,
                              hipStream_t stream) {
    (void)in_sizes; (void)n_in; (void)out_size; (void)ws_size;
    const float* q    = (const float*)d_in[0];
    const float* k    = (const float*)d_in[1];
    const float* v    = (const float*)d_in[2];
    const float* beta = (const float*)d_in[3];
    const int*   mask = (const int*)d_in[4];
    const float* Wq   = (const float*)d_in[5];
    const float* bq   = (const float*)d_in[6];
    const float* Wk   = (const float*)d_in[7];
    const float* bk   = (const float*)d_in[8];
    const float* Wv   = (const float*)d_in[9];
    const float* bv   = (const float*)d_in[10];
    const float* Wo   = (const float*)d_in[11];
    const float* bo   = (const float*)d_in[12];
    float* out = (float*)d_out;

    const size_t elems = (size_t)BATCH * N_HEADS * SEQ * D_HEAD;  // 8M elems
    bf16_t* qh = (bf16_t*)d_ws;          // 16 MB each region
    bf16_t* kh = qh + elems;
    bf16_t* vt = kh + elems;             // V stored transposed per head
    bf16_t* at = vt + elems;

    dim3 gp(BATCH * SEQ / 64, D_MODEL / 64, 3);
    qkv_proj_kernel<<<gp, 128, 0, stream>>>(q, k, v, Wq, Wk, Wv, bq, bk, bv,
                                            qh, kh, vt);

    dim3 ga(BATCH * N_HEADS, SEQ / 64, 1);
    attn_kernel<<<ga, 128, 0, stream>>>(qh, kh, vt, beta, mask, at);

    dim3 go(BATCH * SEQ / 64, D_MODEL / 64, 1);
    out_proj_kernel<<<go, 128, 0, stream>>>(at, Wo, bo, out);
}
